// EnhancedRealityStoneLinear_29068338659717
// MI455X (gfx1250) — matrix operand order repaired and checked by reading, not compiled, not run on hardware
//
#include <hip/hip_runtime.h>
#include <stdint.h>

typedef __attribute__((ext_vector_type(16))) _Float16 v16h;
typedef __attribute__((ext_vector_type(8)))  float    v8f;

#define TOKENS 8192
#define IN_F   4096
#define OUT_F  4096
#define RANK   512

#define BM 128
#define BN 128
#define BK 32
#define LDSB 40                    // _Float16 per LDS row: 80 B = 5*16B chunks
#define BUF_HALVES (BM * LDSB)     // 5120 halves = 10240 B per buffer
#define BUF_BYTES  (BUF_HALVES * 2)
#define DEPTH 4                    // pipeline depth (LDS: DEPTH*20KB = 80KB)

// ---------------- CDNA5 async global->LDS copy (ASYNCcnt path) ----------------
// Copies 16 bytes per lane directly into LDS, no VGPR round trip.
// GVS addressing: mem = SGPR base + 32-bit per-lane byte offset.
__device__ __forceinline__ void async_copy16(uint32_t lds_addr, uint64_t base, uint32_t goff)
{
    asm volatile("global_load_async_to_lds_b128 %0, %1, %2"
                 :
                 : "v"(lds_addr), "v"(goff), "s"(base)
                 : "memory");
}

// Wait until at most N async ops remain in flight (ASYNCcnt <= N).
// Async loads complete in order, so N = 4*(DEPTH-2) leaves the two newest
// tiles in flight while guaranteeing the tile we consume next has landed.
#define WAIT_ASYNC(n) asm volatile("s_wait_asynccnt %0" :: "i"(n) : "memory")

// LDS byte offset of a shared-memory pointer (generic-addr low 32 bits == LDS offset).
__device__ __forceinline__ uint32_t lds_offset(const void* p)
{
    return (uint32_t)(uintptr_t)p;
}

// ---------------- WMMA fragment loads from LDS ----------------
// A fragment (16x32 f16, M rows): lanes 0-15 row M=lane, K {0..7,16..23};
// lanes 16-31 same rows, K {8..15,24..31}.   (ISA 7.12.2, 16-bit A 16x32)
__device__ __forceinline__ v16h load_frag_a(const _Float16* __restrict__ lds,
                                            int rowBase, int lane)
{
    const int m = rowBase + (lane & 15);
    const int h = (lane >> 4) & 1;
    const _Float16* p = lds + m * LDSB;
    union { v16h v; uint4 q[2]; } f;
    f.q[0] = *(const uint4*)(p + h * 8);        // K 0..7   / 8..15
    f.q[1] = *(const uint4*)(p + 16 + h * 8);   // K 16..23 / 24..31
    return f.v;
}

// B fragment (32x16 f16, K x N): source buffer is [n][k] row-major.
// Lane column N = lane&15; lanes 0-15 K {0..7,16..23}, lanes 16-31 K {8..15,24..31}.
__device__ __forceinline__ v16h load_frag_b(const _Float16* __restrict__ lds,
                                            int colBase, int lane)
{
    const int n = colBase + (lane & 15);
    const int h = (lane >> 4) & 1;
    const _Float16* p = lds + n * LDSB;
    union { v16h v; uint4 q[2]; } f;
    f.q[0] = *(const uint4*)(p + h * 8);
    f.q[1] = *(const uint4*)(p + 16 + h * 8);
    return f.v;
}

// ---------------- depth-DEPTH async-fed WMMA main loop ----------------
// Computes acc += A[BM x K] * B[BN x K]^T for this block's tile, streaming
// K in BK steps through DEPTH LDS buffers fed by global_load_async_to_lds_b128.
// Loop is fissioned into a branch-free steady section and a drain tail.
__device__ __forceinline__ void wmma_pipeline(
    v8f (&acc)[2][4],
    const _Float16* __restrict__ Aglob, int ldA, int rowA0,
    const _Float16* __restrict__ Bglob, int ldB, int rowB0,
    int K,
    _Float16* __restrict__ As,   // [DEPTH][BUF_HALVES]
    _Float16* __restrict__ Bs,   // [DEPTH][BUF_HALVES]
    int tid, int lane, int waveM, int waveN)
{
    // Per-thread chunk assignment: chunk i = tid and i = tid+256.
    //   i -> row r = i>>2, 16B chunk c = i&3; chunk2 is (r+64, same c).
    const int r = tid >> 2;
    const int c = tid & 3;
    const uint32_t ldsRow2 = 64u * (LDSB * 2);           // +64 rows, in LDS bytes
    const uint32_t ldsA0 = lds_offset(As) + (uint32_t)(r * (LDSB * 2) + c * 16);
    const uint32_t ldsB0 = lds_offset(Bs) + (uint32_t)(r * (LDSB * 2) + c * 16);
    const uint32_t goffA = (uint32_t)(((size_t)(rowA0 + r) * ldA + c * 8) * 2);
    const uint32_t goffB = (uint32_t)(((size_t)(rowB0 + r) * ldB + c * 8) * 2);
    const uint32_t gstepA = (uint32_t)((size_t)64 * ldA * 2);
    const uint32_t gstepB = (uint32_t)((size_t)64 * ldB * 2);
    const uint64_t baseA = (uint64_t)Aglob;
    const uint64_t baseB = (uint64_t)Bglob;

    // Issue 4 async b128 ops covering one (A,B) tile pair for K-step kb bytes,
    // into pipeline slot dstOff.
    auto fill_tile = [&](uint32_t dstOff, uint32_t kb) {
        async_copy16(ldsA0 + dstOff,           baseA, goffA + kb);
        async_copy16(ldsA0 + dstOff + ldsRow2, baseA, goffA + kb + gstepA);
        async_copy16(ldsB0 + dstOff,           baseB, goffB + kb);
        async_copy16(ldsB0 + dstOff + ldsRow2, baseB, goffB + kb + gstepB);
    };

    // 12 ds_load_b128 + 8 WMMA on pipeline slot bufIdx.
    auto compute_tile = [&](int bufIdx) {
        const _Float16* At = As + bufIdx * BUF_HALVES;
        const _Float16* Bt = Bs + bufIdx * BUF_HALVES;
        v16h a[2], b[4];
        a[0] = load_frag_a(At, waveM * 32, lane);
        a[1] = load_frag_a(At, waveM * 32 + 16, lane);
        #pragma unroll
        for (int ni = 0; ni < 4; ++ni)
            b[ni] = load_frag_b(Bt, waveN * 64 + ni * 16, lane);
        #pragma unroll
        for (int mi = 0; mi < 2; ++mi)
            #pragma unroll
            for (int ni = 0; ni < 4; ++ni)
                acc[mi][ni] = __builtin_amdgcn_wmma_f32_16x16x32_f16(
                    false, a[mi], false, b[ni], (short)0, acc[mi][ni], false, false);
    };

    // Prologue: issue tiles 0 .. DEPTH-2.
    #pragma unroll
    for (int t = 0; t < DEPTH - 1; ++t)
        fill_tile((uint32_t)(t * BUF_BYTES), (uint32_t)(t * BK * 2));

    const int NK = K / BK;
    int steady = NK - (DEPTH - 1);
    if (steady < 0) steady = 0;

    int buf = 0;
    int kt = 0;

    // Steady state: branch-free body; every iteration waits for the oldest
    // in-flight tile and refills the buffer freed last iteration.
    for (; kt < steady; ++kt) {
        WAIT_ASYNC(4 * (DEPTH - 2));   // tile kt resident (in-order completion)
        __syncthreads();               // ..for all waves; reads of (kt-1)%D done
        const int nt = kt + (DEPTH - 1);
        fill_tile((uint32_t)((nt % DEPTH) * BUF_BYTES),
                  (uint32_t)(nt * BK * 2));   // overlaps WMMAs below
        compute_tile(buf);
        buf = (buf + 1 == DEPTH) ? 0 : buf + 1;
    }

    // Drain: last DEPTH-1 tiles, nothing left to issue.
    for (; kt < NK; ++kt) {
        WAIT_ASYNC(0);
        __syncthreads();
        compute_tile(buf);
        buf = (buf + 1 == DEPTH) ? 0 : buf + 1;
    }

    // Protect LDS reuse by the caller (e.g. phase 2 of the fused kernel).
    __syncthreads();
}

// ---------------- conversion / prep kernels ----------------

// x (fp32) -> f16, plus per-token row sum (for the min_val rank-1 term).
__global__ __launch_bounds__(256)
void k_convert_x(const float* __restrict__ x, _Float16* __restrict__ xh,
                 float* __restrict__ rowsum)
{
    const int row = blockIdx.x;
    const float4* src = (const float4*)(x + (size_t)row * IN_F);
    _Float16* dst = xh + (size_t)row * IN_F;
    float s = 0.f;
    for (int i = threadIdx.x; i < IN_F / 4; i += 256) {
        float4 v = src[i];
        s += v.x + v.y + v.z + v.w;
        union { _Float16 h[4]; uint2 u; } pk;
        pk.h[0] = (_Float16)v.x; pk.h[1] = (_Float16)v.y;
        pk.h[2] = (_Float16)v.z; pk.h[3] = (_Float16)v.w;
        *(uint2*)(dst + i * 4) = pk.u;
    }
    __shared__ float red[256];
    red[threadIdx.x] = s;
    __syncthreads();
    for (int off = 128; off > 0; off >>= 1) {
        if (threadIdx.x < off) red[threadIdx.x] += red[threadIdx.x + off];
        __syncthreads();
    }
    if (threadIdx.x == 0) rowsum[row] = red[0];
}

// q (int32, values 0..255 -> exact in f16) -> f16
__global__ __launch_bounds__(256)
void k_convert_q(const int* __restrict__ q, _Float16* __restrict__ qh, int n4)
{
    const int stride = gridDim.x * blockDim.x;
    for (int i = blockIdx.x * blockDim.x + threadIdx.x; i < n4; i += stride) {
        int4 v = ((const int4*)q)[i];
        union { _Float16 h[4]; uint2 u; } pk;
        pk.h[0] = (_Float16)(float)v.x; pk.h[1] = (_Float16)(float)v.y;
        pk.h[2] = (_Float16)(float)v.z; pk.h[3] = (_Float16)(float)v.w;
        ((uint2*)qh)[i] = pk.u;
    }
}

// generic fp32 -> f16 (used for U, which is already [out_f][rank] = [n][k])
__global__ __launch_bounds__(256)
void k_convert_f(const float* __restrict__ src, _Float16* __restrict__ dst, int n4)
{
    const int stride = gridDim.x * blockDim.x;
    for (int i = blockIdx.x * blockDim.x + threadIdx.x; i < n4; i += stride) {
        float4 v = ((const float4*)src)[i];
        union { _Float16 h[4]; uint2 u; } pk;
        pk.h[0] = (_Float16)v.x; pk.h[1] = (_Float16)v.y;
        pk.h[2] = (_Float16)v.z; pk.h[3] = (_Float16)v.w;
        ((uint2*)dst)[i] = pk.u;
    }
}

// V [in_f][rank] -> Vᵀ f16 [rank][in_f] (LDS-tiled 32x32 transpose)
__global__ __launch_bounds__(256)
void k_transpose_v(const float* __restrict__ V, _Float16* __restrict__ vh)
{
    __shared__ float t[32][33];
    const int i0 = blockIdx.x * 32;   // IN_F direction
    const int r0 = blockIdx.y * 32;   // RANK direction
    #pragma unroll
    for (int j = 0; j < 4; ++j)
        t[threadIdx.y + j * 8][threadIdx.x] =
            V[(size_t)(i0 + threadIdx.y + j * 8) * RANK + r0 + threadIdx.x];
    __syncthreads();
    #pragma unroll
    for (int j = 0; j < 4; ++j)
        vh[(size_t)(r0 + threadIdx.y + j * 8) * IN_F + i0 + threadIdx.x] =
            (_Float16)t[threadIdx.x][threadIdx.y + j * 8];
}

// ---------------- WMMA GEMM 1: Y = (x @ V) * S  (f16 output) ----------------
__global__ __launch_bounds__(256)
void k_svd_gemm(const _Float16* __restrict__ xh,   // [TOKENS][IN_F]
                const _Float16* __restrict__ vh,   // [RANK][IN_F]  (= Vᵀ)
                const float* __restrict__ S,       // [RANK]
                _Float16* __restrict__ yh)         // [TOKENS][RANK]
{
    __shared__ _Float16 As[DEPTH * BUF_HALVES];
    __shared__ _Float16 Bs[DEPTH * BUF_HALVES];

    const int tid  = threadIdx.x;
    const int lane = tid & 31;
    const int wave = tid >> 5;
    const int waveM = wave & 3;        // 4 waves along M
    const int waveN = wave >> 2;       // 2 waves along N
    const int blockM = blockIdx.x * BM;
    const int blockN = blockIdx.y * BN;

    v8f acc[2][4];
    #pragma unroll
    for (int mi = 0; mi < 2; ++mi)
        #pragma unroll
        for (int ni = 0; ni < 4; ++ni)
            acc[mi][ni] = (v8f){0.f, 0.f, 0.f, 0.f, 0.f, 0.f, 0.f, 0.f};

    wmma_pipeline(acc, xh, IN_F, blockM, vh, IN_F, blockN, IN_F,
                  As, Bs, tid, lane, waveM, waveN);

    const int h  = lane >> 4;
    const int lm = lane & 15;
    #pragma unroll
    for (int ni = 0; ni < 4; ++ni) {
        const int n = blockN + waveN * 64 + ni * 16 + lm;
        const float sv = S[n];
        #pragma unroll
        for (int mi = 0; mi < 2; ++mi)
            #pragma unroll
            for (int v = 0; v < 8; ++v) {
                const int m = blockM + waveM * 32 + mi * 16 + h * 8 + v;
                yh[(size_t)m * RANK + n] = (_Float16)(acc[mi][ni][v] * sv);
            }
    }
}

// ---- WMMA GEMM 2 (fused): out = scale*(x@qᵀ) + Y@Uᵀ + min_val*rowsum ----
__global__ __launch_bounds__(256)
void k_fused_gemm(const _Float16* __restrict__ xh,  // [TOKENS][IN_F]
                  const _Float16* __restrict__ qh,  // [OUT_F][IN_F]
                  const _Float16* __restrict__ yh,  // [TOKENS][RANK]
                  const _Float16* __restrict__ uh,  // [OUT_F][RANK]
                  const float* __restrict__ rowsum, // [TOKENS]
                  const float* __restrict__ scale_p,
                  const float* __restrict__ minv_p,
                  float* __restrict__ out)          // [TOKENS][OUT_F]
{
    __shared__ _Float16 As[DEPTH * BUF_HALVES];
    __shared__ _Float16 Bs[DEPTH * BUF_HALVES];

    const int tid  = threadIdx.x;
    const int lane = tid & 31;
    const int wave = tid >> 5;
    const int waveM = wave & 3;
    const int waveN = wave >> 2;
    const int blockM = blockIdx.x * BM;
    const int blockN = blockIdx.y * BN;

    const float sc = *scale_p;
    const float mv = *minv_p;

    v8f acc[2][4];
    #pragma unroll
    for (int mi = 0; mi < 2; ++mi)
        #pragma unroll
        for (int ni = 0; ni < 4; ++ni)
            acc[mi][ni] = (v8f){0.f, 0.f, 0.f, 0.f, 0.f, 0.f, 0.f, 0.f};

    // Phase 1: quantized-weight GEMM, K = IN_F
    wmma_pipeline(acc, xh, IN_F, blockM, qh, IN_F, blockN, IN_F,
                  As, Bs, tid, lane, waveM, waveN);

    // Apply dequant scale, keep accumulating the low-rank path on top.
    #pragma unroll
    for (int mi = 0; mi < 2; ++mi)
        #pragma unroll
        for (int ni = 0; ni < 4; ++ni)
            acc[mi][ni] *= sc;

    // Phase 2: low-rank GEMM, K = RANK
    wmma_pipeline(acc, yh, RANK, blockM, uh, RANK, blockN, RANK,
                  As, Bs, tid, lane, waveM, waveN);

    // Epilogue: + min_val * rowsum[t], store fp32.
    const int h  = lane >> 4;
    const int lm = lane & 15;
    #pragma unroll
    for (int mi = 0; mi < 2; ++mi)
        #pragma unroll
        for (int v = 0; v < 8; ++v) {
            const int m = blockM + waveM * 32 + mi * 16 + h * 8 + v;
            const float bias = mv * rowsum[m];
            float* o = out + (size_t)m * OUT_F + blockN + waveN * 64 + lm;
            #pragma unroll
            for (int ni = 0; ni < 4; ++ni)
                o[ni * 16] = acc[mi][ni][v] + bias;
        }
}

// ---------------- launcher ----------------
extern "C" void kernel_launch(void* const* d_in, const int* in_sizes, int n_in,
                              void* d_out, int out_size, void* d_ws, size_t ws_size,
                              hipStream_t stream)
{
    const float* x     = (const float*)d_in[0];
    const int*   q     = (const int*)  d_in[1];
    const float* scale = (const float*)d_in[2];
    const float* minv  = (const float*)d_in[3];
    const float* U     = (const float*)d_in[4];
    const float* S     = (const float*)d_in[5];
    const float* V     = (const float*)d_in[6];
    float* out = (float*)d_out;

    char* ws = (char*)d_ws;
    size_t off = 0;
    _Float16* xh = (_Float16*)(ws + off); off += (size_t)TOKENS * IN_F  * 2;  // 64 MiB
    _Float16* qh = (_Float16*)(ws + off); off += (size_t)OUT_F  * IN_F  * 2;  // 32 MiB
    _Float16* vh = (_Float16*)(ws + off); off += (size_t)RANK   * IN_F  * 2;  //  4 MiB
    _Float16* uh = (_Float16*)(ws + off); off += (size_t)OUT_F  * RANK  * 2;  //  4 MiB
    _Float16* yh = (_Float16*)(ws + off); off += (size_t)TOKENS * RANK  * 2;  //  8 MiB
    float* rowsum = (float*)(ws + off);   off += (size_t)TOKENS * 4;

    // Precision prep (one-time, bandwidth-trivial vs. the 344-GFLOP GEMMs).
    k_convert_x<<<TOKENS, 256, 0, stream>>>(x, xh, rowsum);
    k_convert_q<<<4096, 256, 0, stream>>>(q, qh, (OUT_F * IN_F) / 4);
    k_convert_f<<<2048, 256, 0, stream>>>(U, uh, (OUT_F * RANK) / 4);
    k_transpose_v<<<dim3(IN_F / 32, RANK / 32), dim3(32, 8), 0, stream>>>(V, vh);

    // Y = (x @ V) * S
    k_svd_gemm<<<dim3(TOKENS / BM, RANK / BN), 256, 0, stream>>>(xh, vh, S, yh);

    // out = scale*(x@qᵀ) + Y@Uᵀ + min_val*rowsum
    k_fused_gemm<<<dim3(TOKENS / BM, OUT_F / BN), 256, 0, stream>>>(
        xh, qh, yh, uh, rowsum, scale, minv, out);
}
